// Model_68152541052991
// MI455X (gfx1250) — matrix operand order repaired and checked by reading, not compiled, not run on hardware
//
#include <hip/hip_runtime.h>

// ---------------------------------------------------------------------------
// Paged chunked-prefill causal attention for MI455X (gfx1250), wave32 WMMA.
//   B=4 seqs, Lq=512, S=2048, Hq=32, Kvh=8 (GQA G=4), D=128, page=32 tokens.
// One wave per (seq, q-head, 16-query tile). K/V are L2-resident (33.6MB).
// S^T = K x Q^T via v_wmma_f32_16x16x32_f16, in-lane online softmax,
// O = P x V with P repacked in-lane. V pages are streamed into LDS by the
// Tensor Data Mover (tensor_load_to_lds, TENSORcnt), double-buffered, with
// D# padding (2 dwords / 128) for bank-conflict-free transposed reads.
// ---------------------------------------------------------------------------

#define NUM_SEQS      4
#define Q_LEN         512
#define NUM_Q_HEADS   32
#define NUM_KV_HEADS  8
#define GQA           4
#define HEAD          128
#define BLK           32
#define BLOCKS_PER_SEQ 64
#define NTILES        (Q_LEN / 16)     // 32 q-tiles per (seq, head)
#define VROW          130              // padded LDS row stride (dwords)

typedef __attribute__((ext_vector_type(16))) _Float16 v16h;
typedef __attribute__((ext_vector_type(2)))  _Float16 h2;
typedef __attribute__((ext_vector_type(8)))  float    v8f;
typedef __attribute__((ext_vector_type(4)))  float    v4f;
typedef __attribute__((ext_vector_type(2)))  unsigned int u2;

// 32-byte fragment viewed as v16h (WMMA A/B) or 8 half2.
union AFrag {
    v16h v;
    h2 h[8];
};

__device__ inline h2 pk(float a, float b) {
    return __builtin_bit_cast(h2, __builtin_amdgcn_cvt_pkrtz(a, b));
}
__device__ inline float xor16(float x) {       // swap lanes n <-> n+16
    int i = __builtin_amdgcn_ds_swizzle(__builtin_bit_cast(int, x), 0x401F);
    return __builtin_bit_cast(float, i);
}
__device__ inline float lanebc(float x, int src) {  // per-lane broadcast
    int i = __builtin_amdgcn_ds_bpermute(src << 2, __builtin_bit_cast(int, x));
    return __builtin_bit_cast(float, i);
}

#if __has_builtin(__builtin_amdgcn_tensor_load_to_lds)
#define HAVE_TDM 1
typedef unsigned int u32x4 __attribute__((ext_vector_type(4)));
typedef int          i32x8 __attribute__((ext_vector_type(8)));
typedef int          i32x4 __attribute__((ext_vector_type(4)));

// DMA one V page (32 rows x 128 f32, row stride 1024 f32) into LDS at
// lds_off, with 2 pad dwords inserted every 128 dwords (row stride 130).
__device__ inline void tdm_load_v(const float* gptr, unsigned lds_off) {
    const unsigned long long ga = (unsigned long long)(const void*)gptr;
    u32x4 g0;
    g0[0] = 1u;                                   // count=1 (valid), user mode
    g0[1] = lds_off;                              // LDS byte address
    g0[2] = (unsigned)(ga & 0xffffffffu);         // global_addr[31:0]
    g0[3] = (unsigned)((ga >> 32) & 0x1ffffffu)   // global_addr[56:32]
          | (2u << 30);                           // type = 2 ("image")
    i32x8 g1;
    g1[0] = (2 << 16)      // data_size = 4B
          | (1 << 20)      // pad_enable
          | (6 << 22)      // pad_interval: every 128 dwords
          | (1 << 25);     // pad_amount: 2 dwords
    g1[1] = (int)(128u << 16);   // tensor_dim0 = 128 (bits 79:48, low half)
    g1[2] = (int)(32u  << 16);   // tensor_dim1 = 32  (bits 111:80, low half)
    g1[3] = (int)(128u << 16);   // tile_dim0 = 128   (bits 127:112)
    g1[4] = 32;                  // tile_dim1 = 32, tile_dim2 = 0
    g1[5] = 1024;                // tensor_dim0_stride = 1024 elements
    g1[6] = 0;
    g1[7] = 0;
    const i32x4 z4 = {0, 0, 0, 0};                // groups 2/3: unused (2-D)
    const i32x8 z8 = {0, 0, 0, 0, 0, 0, 0, 0};    // extra group (clang-23 form)
    __asm__ __volatile__("" ::: "memory");
    __builtin_amdgcn_tensor_load_to_lds(g0, g1, z4, z4, z8, 0);
    __asm__ __volatile__("" ::: "memory");
}
#else
#define HAVE_TDM 0
#endif

// A-fragment of one 16x32 f16 K tile (M=16 keys, K=32 head-dims).
// ISA 7.12.2: lane m    : elems 0-7 -> d=c*32+0..7,  8-15 -> d=c*32+16..23
//             lane m+16 : elems 0-7 -> d=c*32+8..15, 8-15 -> d=c*32+24..31
__device__ inline void load_kfrag(AFrag& a, const float* __restrict__ krow,
                                  int c, int hi) {
    const int d0 = c * 32 + hi * 8;
    const v4f f0 = *(const v4f*)(krow + d0);
    const v4f f1 = *(const v4f*)(krow + d0 + 4);
    const v4f f2 = *(const v4f*)(krow + d0 + 16);
    const v4f f3 = *(const v4f*)(krow + d0 + 20);
    a.h[0] = pk(f0.x, f0.y); a.h[1] = pk(f0.z, f0.w);
    a.h[2] = pk(f1.x, f1.y); a.h[3] = pk(f1.z, f1.w);
    a.h[4] = pk(f2.x, f2.y); a.h[5] = pk(f2.z, f2.w);
    a.h[6] = pk(f3.x, f3.y); a.h[7] = pk(f3.z, f3.w);
}

__global__ __launch_bounds__(32)
void paged_prefill_attn(const float* __restrict__ q,
                        const float* __restrict__ kc,
                        const float* __restrict__ vc,
                        const int*   __restrict__ bt,
                        const int*   __restrict__ slen,
                        const int*   __restrict__ qsl,
                        float*       __restrict__ out) {
    // Double-buffered V page staging, fp32, padded rows (130 dwords).
    __shared__ float VL[2][32 * VROW];            // 33280 B

    const int bx   = blockIdx.x;
    const int tile = bx & (NTILES - 1);
    const int h    = (bx >> 5) & (NUM_Q_HEADS - 1);
    const int b    = bx >> 10;

    const int lane = threadIdx.x;
    const int n    = lane & 15;        // query column (and key-row index)
    const int hi   = lane >> 4;        // lane half
    const int kvh  = h / GQA;

    const int qstart = qsl[b];
    const int seqlen = slen[b];
    const int ctx    = seqlen - Q_LEN;
    const int l0     = tile * 16;
    const int q_abs  = ctx + l0 + n;   // this lane's absolute query position

    // softmax in exp2 domain: fold 1/sqrt(D) * log2(e) into Q.
    const float sc = 0.08838834764831845f * 1.4426950408889634f;

    // ---- Q^T B-fragments (16 queries x 128 d), loaded once, pre-scaled ----
    // 16-bit B layout (AMD table order): VGPR v halves = rows 2v,2v+1 (+8 if v>=4); lane half = +8.
    // => lane n, elem e  <->  d = c*32 + hi*8 + e (e<8), c*32 + 16 + hi*8 + e-8 (e>=8).
    AFrag qf[4];
    {
        const float* qp = q + ((size_t)(qstart + l0 + n) * NUM_Q_HEADS + h) * HEAD;
#pragma unroll
        for (int c = 0; c < 4; ++c) {
            const int d0 = c * 32 + hi * 8;
#pragma unroll
            for (int j = 0; j < 4; ++j) {
                const v4f f = *(const v4f*)(qp + d0 + 4 * j + (j & 2) * 4);
                qf[c].h[2 * j]     = pk(f.x * sc, f.y * sc);
                qf[c].h[2 * j + 1] = pk(f.z * sc, f.w * sc);
            }
        }
    }

    v8f o[8];
#pragma unroll
    for (int t = 0; t < 8; ++t) o[t] = v8f{0.f, 0.f, 0.f, 0.f, 0.f, 0.f, 0.f, 0.f};
    float m_run = -1e30f, l_run = 0.0f;

    const int nkeys   = ctx + l0 + 16;              // causal horizon of tile
    const int nchunks = (nkeys + BLK - 1) / BLK;    // one chunk == one page
    const int* btb    = bt + b * BLOCKS_PER_SEQ;

    const size_t kvrow = (size_t)NUM_KV_HEADS * HEAD;   // 1024 floats

#if HAVE_TDM
    // Prologue: start DMA of V page 0 into buffer 0.
    {
        const int blk0 = btb[0];
        tdm_load_v(vc + ((size_t)blk0 * BLK * NUM_KV_HEADS + kvh) * HEAD,
                   (unsigned)(size_t)(void*)&VL[0][0]);
    }
#endif

    for (int ck = 0; ck < nchunks; ++ck) {
        const int blk = btb[ck];
        const float* kb = kc + (((size_t)blk * BLK) * NUM_KV_HEADS + kvh) * HEAD;

        // ---- kick off DMA of the NEXT V page into the other buffer ----
        const int nk   = (ck + 1 < nchunks) ? (ck + 1) : (nchunks - 1);
        const int nblk = btb[nk];
        const float* vnext =
            vc + (((size_t)nblk * BLK) * NUM_KV_HEADS + kvh) * HEAD;
#if HAVE_TDM
        tdm_load_v(vnext, (unsigned)(size_t)(void*)&VL[(ck + 1) & 1][0]);
#endif

        // ---- S^T = K x Q^T : two 16-key tiles, 4 d-chunks of 32 each ----
        v8f st0 = v8f{0.f, 0.f, 0.f, 0.f, 0.f, 0.f, 0.f, 0.f};
        v8f st1 = st0;
        const float* krow0 = kb + (size_t)n        * kvrow;
        const float* krow1 = kb + (size_t)(n + 16) * kvrow;
#pragma unroll
        for (int c = 0; c < 4; ++c) {
            AFrag ka0, ka1;
            load_kfrag(ka0, krow0, c, hi);
            load_kfrag(ka1, krow1, c, hi);
            st0 = __builtin_amdgcn_wmma_f32_16x16x32_f16(
                false, ka0.v, false, qf[c].v, (short)0, st0, false, false);
            st1 = __builtin_amdgcn_wmma_f32_16x16x32_f16(
                false, ka1.v, false, qf[c].v, (short)0, st1, false, false);
        }
        // st0: lane col = query n; vgpr j = key j (+8 for hi). st1 keys +16.

        // ---- causal mask (only boundary chunks; uniform branch) ----
        if (ck * BLK + (BLK - 1) > ctx + l0) {
#pragma unroll
            for (int j = 0; j < 8; ++j) {
                const int k0 = ck * BLK + j + hi * 8;
                st0[j] = (k0 > q_abs)      ? -1e30f : st0[j];
                st1[j] = (k0 + 16 > q_abs) ? -1e30f : st1[j];
            }
        }

        // ---- online softmax: per-lane over 16 keys + one xor16 merge ----
        float mloc = st0[0];
#pragma unroll
        for (int j = 1; j < 8; ++j) mloc = fmaxf(mloc, st0[j]);
#pragma unroll
        for (int j = 0; j < 8; ++j) mloc = fmaxf(mloc, st1[j]);
        mloc = fmaxf(mloc, xor16(mloc));
        const float mnew  = fmaxf(m_run, mloc);
        const float alpha = exp2f(m_run - mnew);
        float p0[8], p1[8], ls = 0.f;
#pragma unroll
        for (int j = 0; j < 8; ++j) {
            p0[j] = exp2f(st0[j] - mnew); ls += p0[j];
            p1[j] = exp2f(st1[j] - mnew); ls += p1[j];
        }
        ls += xor16(ls);
        l_run = l_run * alpha + ls;
        m_run = mnew;

        // ---- P A-fragment: direct in-lane repack of S^T tiles ----
        AFrag pf;
#pragma unroll
        for (int j = 0; j < 4; ++j) {
            pf.h[j]     = pk(p0[2 * j], p0[2 * j + 1]);
            pf.h[4 + j] = pk(p1[2 * j], p1[2 * j + 1]);
        }

        // ---- rescale O rows by alpha (row j lives in lane j (+8 hi)) ----
        float av[8];
#pragma unroll
        for (int j = 0; j < 8; ++j) av[j] = lanebc(alpha, j + hi * 8);
#pragma unroll
        for (int t = 0; t < 8; ++t)
#pragma unroll
            for (int j = 0; j < 8; ++j) o[t][j] *= av[j];

        // ---- make current V page visible, then O += P x V ----
        float* VLb = &VL[ck & 1][0];
#if HAVE_TDM
        // TDM ops complete in order: <=1 outstanding leaves only next page.
        __builtin_amdgcn_s_wait_tensorcnt(1);
        __asm__ __volatile__("" ::: "memory");
#else
        // Fallback staging: lane covers row r, cols 4*lane..4*lane+3.
        const float* vb = vc + (((size_t)blk * BLK) * NUM_KV_HEADS + kvh) * HEAD;
        (void)vnext;
#pragma unroll
        for (int r = 0; r < 32; ++r) {
            const v4f f = *(const v4f*)(vb + (size_t)r * kvrow + lane * 4);
            u2 lo, hiw;
            lo.x  = __builtin_bit_cast(unsigned int, f.x);
            lo.y  = __builtin_bit_cast(unsigned int, f.y);
            hiw.x = __builtin_bit_cast(unsigned int, f.z);
            hiw.y = __builtin_bit_cast(unsigned int, f.w);
            *(u2*)(&VLb[r * VROW + lane * 4])     = lo;
            *(u2*)(&VLb[r * VROW + lane * 4 + 2]) = hiw;
        }
#endif

        // V B-fragment: VGPR v = half2{ V[r][d], V[r+1][d] }, r = 2v + 8*hi (+8 for
        // v>=4); reads pair to ds_load_2addr (offset1 = 130).
#pragma unroll
        for (int t = 0; t < 8; ++t) {
            AFrag vf;
#pragma unroll
            for (int v = 0; v < 8; ++v) {
                const float* vl = &VLb[(2 * v + hi * 8 + (v & 4) * 2) * VROW + t * 16 + n];
                vf.h[v] = pk(vl[0], vl[VROW]);
            }
            o[t] = __builtin_amdgcn_wmma_f32_16x16x32_f16(
                false, pf.v, false, vf.v, (short)0, o[t], false, false);
        }
    }

    // ---- epilogue: divide by row sums, store fp32 ----
    float linv[8];
#pragma unroll
    for (int j = 0; j < 8; ++j)
        linv[j] = __builtin_amdgcn_rcpf(lanebc(l_run, j + hi * 8));

    float* op = out + ((size_t)(qstart + l0) * NUM_Q_HEADS + h) * HEAD;
#pragma unroll
    for (int t = 0; t < 8; ++t)
#pragma unroll
        for (int j = 0; j < 8; ++j)
            op[(size_t)(j + hi * 8) * NUM_Q_HEADS * HEAD + t * 16 + n] =
                o[t][j] * linv[j];
}

extern "C" void kernel_launch(void* const* d_in, const int* in_sizes, int n_in,
                              void* d_out, int out_size, void* d_ws, size_t ws_size,
                              hipStream_t stream) {
    const float* q   = (const float*)d_in[0];
    const float* kc  = (const float*)d_in[1];
    const float* vc  = (const float*)d_in[2];
    const int*   bt  = (const int*)d_in[3];
    const int*   sl  = (const int*)d_in[4];
    const int*   qsl = (const int*)d_in[5];
    float* out = (float*)d_out;

    const dim3 grid(NUM_SEQS * NUM_Q_HEADS * NTILES);  // 4096 waves
    paged_prefill_attn<<<grid, 32, 0, stream>>>(q, kc, vc, bt, sl, qsl, out);
}
